// GAT_79431125172706
// MI455X (gfx1250) — matrix operand order repaired and checked by reading, not compiled, not run on hardware
//
#include <hip/hip_runtime.h>
#include <hip/hip_bf16.h>

// ---------------------------------------------------------------------------
// 3-layer GAT (ogbn-products style) for gfx1250 (MI455X).
//   - Activations/weights packed to bf16, K padded to 128, M padded to 16:
//     zero guards in the WMMA inner loop.
//   - Weights staged per workgroup into LDS by the Tensor Data Mover
//     (tensor_load_to_lds + s_wait_tensorcnt), falling back to a cooperative
//     b128 copy if the builtin is unavailable.
//   - B fragments from ds_load_b128; all four A fragments loaded up-front in
//     one global b128 clause; one wave computes 16 rows x all output columns
//     (A reuse across CT back-to-back v_wmma ops).
//   - combine_relu emits bf16 directly -> no separate activation re-pack pass.
//   - Edge softmax (segment max / sum) via float atomics; aggregation with one
//     wave32 per edge, lane == channel (coalesced gather/scatter).
// ---------------------------------------------------------------------------

#define HEADS 4
#define NEG_SLOPE 0.2f

typedef __attribute__((ext_vector_type(16))) __bf16 v16bf;
typedef __attribute__((ext_vector_type(8)))  float  v8f;
typedef __attribute__((ext_vector_type(4)))  unsigned int v4u;
typedef __attribute__((ext_vector_type(8)))  int    v8i;
typedef __attribute__((ext_vector_type(4)))  int    v4i;

static __device__ __forceinline__ unsigned short bf16_bits(float f) {
  const unsigned u = __float_as_uint(f);
  return (unsigned short)((u + 0x7FFFu + ((u >> 16) & 1u)) >> 16);  // RNE
}

static __device__ __forceinline__ float lrelu(float v) {
  return v > 0.f ? v : NEG_SLOPE * v;
}

// float atomic max via integer atomics (valid for IEEE754 ordering)
static __device__ __forceinline__ void atomicMaxF(float* addr, float v) {
  if (v >= 0.f) atomicMax((int*)addr, __float_as_int(v));
  else          atomicMin((unsigned int*)addr, __float_as_uint(v));
}

// edge id -> (src, dst); edges >= E are the appended self-loops
static __device__ __forceinline__ void edge_nodes(const int* __restrict__ ei,
                                                  int e, int E, int& s, int& d) {
  if (e < E) { s = ei[e]; d = ei[E + e]; } else { s = e - E; d = e - E; }
}

// ---------------------------------------------------------------------------
// Pack fp32 [rowsValid, K] -> bf16-bits [rowsPad, Kp], zero padded.
// ---------------------------------------------------------------------------
__global__ void pack_bf16_kernel(const float* __restrict__ in,
                                 unsigned short* __restrict__ out,
                                 int rowsValid, int rowsPad, int K, int Kp)
{
  const int t = blockIdx.x * blockDim.x + threadIdx.x;
  if (t >= rowsPad * Kp) return;
  const int r = t / Kp, k = t % Kp;
  out[t] = (r < rowsValid && k < K) ? bf16_bits(in[(size_t)r * K + k])
                                    : (unsigned short)0;
}

// ---------------------------------------------------------------------------
// GEMM: C[Nrows, M] = A[Nrows, 128]bf16 @ W[CT*16, 128]bf16^T (+ bias[M])
// Block = 8 waves. W staged into LDS by TDM (one DMA from wave 0); each wave
// owns one 16-row tile and covers all CT column tiles.
// Fragment layouts per CDNA5 ISA 7.12.2 (wave32):
//   A 16x32:  lane%16 -> M row; lane half selects the 8-wide K subgroup;
//             e0..7 -> K = k0+half*8+(0..7), e8..15 -> K = k0+16+half*8+(0..7).
//   B 32x16:  lane%16 -> N col; e0..7 -> K = k0+half*8+(0..7), e8..15 -> K = k0+16+half*8+(0..7).
//   C/D f32:  lane%16 -> N col; VGPR r -> M = half*8 + r.
// ---------------------------------------------------------------------------
template<int CT>
__global__ __launch_bounds__(256)
void gemm_bf16_wmma(const unsigned short* __restrict__ A,   // [Nrows, 128]
                    const unsigned short* __restrict__ W,   // [CT*16, 128]
                    const float* __restrict__ bias,         // [M] or null
                    float* __restrict__ C,                  // [Nrows, M]
                    int Nrows, int M)
{
  constexpr int KP  = 128;
  constexpr int NIT = 4 * CT;                  // (KP/32) * CT fragment steps
  __shared__ unsigned short Wlds[CT * 16 * KP];

  const int tid = threadIdx.x;

#if __has_builtin(__builtin_amdgcn_tensor_load_to_lds)
  // ---- W -> LDS via Tensor Data Mover: one DMA issued by wave 0 -----------
  if ((tid >> 5) == 0) {
    const unsigned lds_off = (unsigned)(size_t)(&Wlds[0]); // LDS byte offset
    const unsigned long long ga = (unsigned long long)(size_t)W;
    // D# group 0: count=1 | lds_addr | global_addr[56:0] | type=2
    v4u g0;
    g0[0] = 0x1u;                                   // count=1, user D#
    g0[1] = lds_off;                                // lds_addr (bytes)
    g0[2] = (unsigned)(ga & 0xFFFFFFFFu);           // global_addr[31:0]
    g0[3] = (unsigned)((ga >> 32) & 0x01FFFFFFu)    // global_addr[56:32]
            | (2u << 30);                           // type = 2 ("image")
    // D# group 1: mask=0, data_size=1 (2B), 2-D tile == whole tensor
    v8i g1;
    g1[0] = 0x00010000;                             // data_size=1, flags=0
    g1[1] = (int)(128u << 16);                      // tensor_dim0 = 128 (lo16)
    g1[2] = (int)((unsigned)(CT * 16) << 16);       // td0 hi=0 | tensor_dim1 lo
    g1[3] = (int)(128u << 16);                      // td1 hi=0 | tile_dim0=128
    g1[4] = CT * 16;                                // tile_dim1 | tile_dim2=0
    g1[5] = 128;                                    // tensor_dim0_stride lo
    g1[6] = 0;                                      // stride hi | dim1_stride lo
    g1[7] = 0;                                      // dim1_stride hi
    v4i g2 = {0, 0, 0, 0};                          // dims 2/3 unused
    v4i g3 = {0, 0, 0, 0};
#if defined(__clang_major__) && (__clang_major__ >= 23)
    v8i g4 = {0, 0, 0, 0, 0, 0, 0, 0};
    __builtin_amdgcn_tensor_load_to_lds(g0, g1, g2, g3, g4, 0);
#else
    __builtin_amdgcn_tensor_load_to_lds(g0, g1, g2, g3, 0);
#endif
    __builtin_amdgcn_s_wait_tensorcnt(0);           // DMA complete in LDS
  }
  __syncthreads();
#else
  // ---- fallback: cooperative W -> LDS staging (b128 granules) -------------
#pragma unroll
  for (int i = 0; i < CT; ++i)
    ((uint4*)Wlds)[i * 256 + tid] = ((const uint4*)W)[i * 256 + tid];
  __syncthreads();
#endif

  const int lane   = tid & 31;
  const int wave   = (blockIdx.x << 3) + (tid >> 5);
  const int tilesN = (Nrows + 15) >> 4;
  const bool valid = (wave < tilesN);
  const int wv     = valid ? wave : (tilesN - 1);  // clamp: all waves compute
  const int m0     = wv << 4;
  const int half   = lane >> 4;
  const int lanelo = lane & 15;

  int arow = m0 + lanelo;
  if (arow >= Nrows) arow = Nrows - 1;
  const unsigned short* Arow = A + (size_t)arow * KP + (half << 3);
  const unsigned short* Wl0  = Wlds + lanelo * KP + (half << 3);

  union BF { uint4 u[2]; v16bf v; };

  // all four A fragments up front (single clause of 8 global_load_b128)
  BF af[4];
#pragma unroll
  for (int kk = 0; kk < 4; ++kk) {
    af[kk].u[0] = *(const uint4*)(Arow + kk * 32);       // e0..7
    af[kk].u[1] = *(const uint4*)(Arow + kk * 32 + 16);  // e8..15
  }

  v8f acc[CT];
  const v8f vz = {0.f, 0.f, 0.f, 0.f, 0.f, 0.f, 0.f, 0.f};
#pragma unroll
  for (int j = 0; j < CT; ++j) acc[j] = vz;

  // ---- software-pipelined B fragments from LDS ----------------------------
  BF bc, bn;
  bc.u[0] = *(const uint4*)(Wl0);
  bc.u[1] = *(const uint4*)(Wl0 + 16);
#pragma unroll
  for (int it = 0; it < NIT; ++it) {
    const int k0 = (it / CT) * 32;
    const int j  = it % CT;
    if (it + 1 < NIT) {                       // prefetch next fragment
      const int k0n = ((it + 1) / CT) * 32;
      const int jn  = (it + 1) % CT;
      const unsigned short* p = Wl0 + (size_t)(jn << 4) * KP + k0n;
      bn.u[0] = *(const uint4*)(p);
      bn.u[1] = *(const uint4*)(p + 16);
    }
    acc[j] = __builtin_amdgcn_wmma_f32_16x16x32_bf16(
        /*neg_a=*/false, af[k0 >> 5].v, /*neg_b=*/false, bc.v,
        /*c_mod=*/(short)0, acc[j], /*reuse_a=*/false, /*reuse_b=*/false);
    if (it + 1 < NIT) bc = bn;                // renamed away by full unroll
  }

  if (valid) {
#pragma unroll
    for (int j = 0; j < CT; ++j) {
      const int col = (j << 4) + lanelo;
      if (col < M) {
        const float bb = bias ? bias[col] : 0.f;
#pragma unroll
        for (int r = 0; r < 8; ++r) {
          const int row = m0 + (half << 3) + r;
          if (row < Nrows) C[(size_t)row * M + col] = acc[j][r] + bb;
        }
      }
    }
  }
}

// alpha_src[n,h] = sum_c h[n,h,c]*a_src[h,c]   (and same for dst)
__global__ void alpha_kernel(const float* __restrict__ h,
                             const float* __restrict__ a_src,
                             const float* __restrict__ a_dst,
                             float* __restrict__ out_src,
                             float* __restrict__ out_dst,
                             int Nn, int C)
{
  const int t = blockIdx.x * blockDim.x + threadIdx.x;
  if (t >= Nn * HEADS) return;
  const int n = t / HEADS, hd = t % HEADS;
  const float* hp = h + ((size_t)n * HEADS + hd) * C;
  const float* as = a_src + hd * C;
  const float* ad = a_dst + hd * C;
  float s = 0.f, d = 0.f;
  for (int c = 0; c < C; ++c) { const float v = hp[c]; s += v * as[c]; d += v * ad[c]; }
  out_src[t] = s;
  out_dst[t] = d;
}

__global__ void fill_kernel(float* __restrict__ p, float v, int n) {
  const int i = blockIdx.x * blockDim.x + threadIdx.x;
  if (i < n) p[i] = v;
}

__global__ void edge_max_kernel(const int* __restrict__ ei, int E, int ET,
                                const float* __restrict__ asrc,
                                const float* __restrict__ adst,
                                float* __restrict__ mx)
{
  const int t = blockIdx.x * blockDim.x + threadIdx.x;
  if (t >= ET * HEADS) return;
  const int e = t / HEADS, hd = t % HEADS;
  int s, d; edge_nodes(ei, e, E, s, d);
  const float v = lrelu(asrc[s * HEADS + hd] + adst[d * HEADS + hd]);
  atomicMaxF(&mx[d * HEADS + hd], v);
}

__global__ void edge_sum_kernel(const int* __restrict__ ei, int E, int ET,
                                const float* __restrict__ asrc,
                                const float* __restrict__ adst,
                                const float* __restrict__ mx,
                                float* __restrict__ den)
{
  const int t = blockIdx.x * blockDim.x + threadIdx.x;
  if (t >= ET * HEADS) return;
  const int e = t / HEADS, hd = t % HEADS;
  int s, d; edge_nodes(ei, e, E, s, d);
  const float v = lrelu(asrc[s * HEADS + hd] + adst[d * HEADS + hd]);
  atomicAdd(&den[d * HEADS + hd], __expf(v - mx[d * HEADS + hd]));
}

// one wave32 per edge; lane == channel chunk -> coalesced gather/scatter
__global__ __launch_bounds__(256)
void edge_agg_kernel(const int* __restrict__ ei, int E, int ET,
                     const float* __restrict__ asrc,
                     const float* __restrict__ adst,
                     const float* __restrict__ mx,
                     const float* __restrict__ den,
                     const float* __restrict__ h,
                     float* __restrict__ agg, int C)
{
  const int wave = (blockIdx.x << 3) + (threadIdx.x >> 5);
  if (wave >= ET) return;
  const int lane = threadIdx.x & 31;
  int s, d; edge_nodes(ei, wave, E, s, d);
  const int HC = HEADS * C;
  for (int ch = lane; ch < HC; ch += 32) {
    const int hd = ch / C;
    const float v = lrelu(asrc[s * HEADS + hd] + adst[d * HEADS + hd]);
    const float a = __expf(v - mx[d * HEADS + hd]) / (den[d * HEADS + hd] + 1e-16f);
    atomicAdd(&agg[(size_t)d * HC + ch], a * h[(size_t)s * HC + ch]);
  }
}

// x_out(bf16) = relu(agg + b + lin)    (layers 1 & 2; feeds next layer's GEMM)
__global__ void combine_relu_kernel(const float* __restrict__ agg,
                                    const float* __restrict__ b,
                                    const float* __restrict__ lin,
                                    unsigned short* __restrict__ out_bf,
                                    int Nn, int HC)
{
  const int t = blockIdx.x * blockDim.x + threadIdx.x;
  if (t >= Nn * HC) return;
  const int ch = t % HC;
  const float v = agg[t] + b[ch] + lin[t];
  out_bf[t] = bf16_bits(v > 0.f ? v : 0.f);
}

// layer 3: mean over heads + b3 + lin, then log_softmax over classes
__global__ void final_kernel(const float* __restrict__ agg,  // [N, H, C]
                             const float* __restrict__ b3,   // [C]
                             const float* __restrict__ lin,  // [N, C]
                             float* __restrict__ out, int Nn, int C)
{
  const int n = blockIdx.x * blockDim.x + threadIdx.x;
  if (n >= Nn) return;
  float tmp[64];
  float mx = -1e30f;
  for (int c = 0; c < C; ++c) {
    float v = 0.f;
    for (int hd = 0; hd < HEADS; ++hd) v += agg[((size_t)n * HEADS + hd) * C + c];
    v = v * (1.0f / HEADS) + b3[c] + lin[(size_t)n * C + c];
    tmp[c] = v;
    if (v > mx) mx = v;
  }
  float se = 0.f;
  for (int c = 0; c < C; ++c) se += __expf(tmp[c] - mx);
  const float lse = mx + __logf(se);
  for (int c = 0; c < C; ++c) out[(size_t)n * C + c] = tmp[c] - lse;
}

// ---------------------------------------------------------------------------
// Host side
// ---------------------------------------------------------------------------
static inline int cdiv(long a, long b) { return (int)((a + b - 1) / b); }

extern "C" void kernel_launch(void* const* d_in, const int* in_sizes, int n_in,
                              void* d_out, int out_size, void* d_ws, size_t ws_size,
                              hipStream_t stream) {
  const int IN = 100, HID = 32, OUT = 47;
  const int KP = 128;                 // padded K for every GEMM
  const int N  = in_sizes[0] / IN;
  const int E  = in_sizes[1] / 2;
  const int ET = E + N;               // with self-loops

  const float* x        = (const float*)d_in[0];
  const int*   ei       = (const int*)d_in[1];
  const float* W1       = (const float*)d_in[2];
  const float* att1_src = (const float*)d_in[3];
  const float* att1_dst = (const float*)d_in[4];
  const float* b1       = (const float*)d_in[5];
  const float* Wl1      = (const float*)d_in[6];
  const float* bl1      = (const float*)d_in[7];
  const float* W2       = (const float*)d_in[8];
  const float* att2_src = (const float*)d_in[9];
  const float* att2_dst = (const float*)d_in[10];
  const float* b2       = (const float*)d_in[11];
  const float* Wl2      = (const float*)d_in[12];
  const float* bl2      = (const float*)d_in[13];
  const float* W3       = (const float*)d_in[14];
  const float* att3_src = (const float*)d_in[15];
  const float* att3_dst = (const float*)d_in[16];
  const float* b3       = (const float*)d_in[17];
  const float* Wl3      = (const float*)d_in[18];
  const float* bl3      = (const float*)d_in[19];
  float* out = (float*)d_out;

  // workspace carve-up
  float* ws   = (float*)d_ws;
  float* h    = ws;                        // N*188 f32 (layers use <= 188 cols)
  float* agg  = h    + (size_t)N * 188;    // N*188 f32
  float* lin  = agg  + (size_t)N * 188;    // N*128 f32 (layer3 uses N*47)
  float* asrc = lin  + (size_t)N * 128;    // N*4 f32
  float* adst = asrc + (size_t)N * HEADS;
  float* mxb  = adst + (size_t)N * HEADS;
  float* den  = mxb  + (size_t)N * HEADS;
  unsigned short* xb = (unsigned short*)(den + (size_t)N * HEADS); // N*128 bf16
  unsigned short* wb = xb + (size_t)N * KP;                        // 192*128 bf16
  (void)ws_size; (void)n_in; (void)out_size;

  const int T = 256;
  const int gemm_blk = cdiv(cdiv(N, 16), 8);       // one wave per 16-row tile
  const int nh_blk   = cdiv((long)N * HEADS, T);
  const int edg_blk  = cdiv((long)ET * HEADS, T);
  const int agg_blk  = cdiv(ET, 8);

  // -------- Layer 1: GATConv(100 -> 32 x 4, concat) + linear skip ----------
  pack_bf16_kernel<<<cdiv((long)N * KP, T), T, 0, stream>>>(x, xb, N, N, IN, KP);
  pack_bf16_kernel<<<cdiv(128L * KP, T), T, 0, stream>>>(W1, wb, 128, 128, IN, KP);
  gemm_bf16_wmma<8><<<gemm_blk, T, 0, stream>>>(xb, wb, nullptr, h, N, 128);
  pack_bf16_kernel<<<cdiv(128L * KP, T), T, 0, stream>>>(Wl1, wb, 128, 128, IN, KP);
  gemm_bf16_wmma<8><<<gemm_blk, T, 0, stream>>>(xb, wb, bl1, lin, N, 128);
  alpha_kernel<<<nh_blk, T, 0, stream>>>(h, att1_src, att1_dst, asrc, adst, N, HID);
  fill_kernel<<<nh_blk, T, 0, stream>>>(mxb, -1e30f, N * HEADS);
  fill_kernel<<<nh_blk, T, 0, stream>>>(den, 0.f, N * HEADS);
  fill_kernel<<<cdiv((long)N * 128, T), T, 0, stream>>>(agg, 0.f, N * 128);
  edge_max_kernel<<<edg_blk, T, 0, stream>>>(ei, E, ET, asrc, adst, mxb);
  edge_sum_kernel<<<edg_blk, T, 0, stream>>>(ei, E, ET, asrc, adst, mxb, den);
  edge_agg_kernel<<<agg_blk, T, 0, stream>>>(ei, E, ET, asrc, adst, mxb, den, h, agg, HID);
  combine_relu_kernel<<<cdiv((long)N * 128, T), T, 0, stream>>>(agg, b1, lin, xb, N, 128);

  // -------- Layer 2: GATConv(128 -> 32 x 4, concat) + linear skip ----------
  pack_bf16_kernel<<<cdiv(128L * KP, T), T, 0, stream>>>(W2, wb, 128, 128, 128, KP);
  gemm_bf16_wmma<8><<<gemm_blk, T, 0, stream>>>(xb, wb, nullptr, h, N, 128);
  pack_bf16_kernel<<<cdiv(128L * KP, T), T, 0, stream>>>(Wl2, wb, 128, 128, 128, KP);
  gemm_bf16_wmma<8><<<gemm_blk, T, 0, stream>>>(xb, wb, bl2, lin, N, 128);
  alpha_kernel<<<nh_blk, T, 0, stream>>>(h, att2_src, att2_dst, asrc, adst, N, HID);
  fill_kernel<<<nh_blk, T, 0, stream>>>(mxb, -1e30f, N * HEADS);
  fill_kernel<<<nh_blk, T, 0, stream>>>(den, 0.f, N * HEADS);
  fill_kernel<<<cdiv((long)N * 128, T), T, 0, stream>>>(agg, 0.f, N * 128);
  edge_max_kernel<<<edg_blk, T, 0, stream>>>(ei, E, ET, asrc, adst, mxb);
  edge_sum_kernel<<<edg_blk, T, 0, stream>>>(ei, E, ET, asrc, adst, mxb, den);
  edge_agg_kernel<<<agg_blk, T, 0, stream>>>(ei, E, ET, asrc, adst, mxb, den, h, agg, HID);
  combine_relu_kernel<<<cdiv((long)N * 128, T), T, 0, stream>>>(agg, b2, lin, xb, N, 128);

  // -------- Layer 3: GATConv(128 -> 47 x 4, mean) + skip + log_softmax -----
  pack_bf16_kernel<<<cdiv(192L * KP, T), T, 0, stream>>>(W3, wb, 188, 192, 128, KP);
  gemm_bf16_wmma<12><<<gemm_blk, T, 0, stream>>>(xb, wb, nullptr, h, N, 188);
  pack_bf16_kernel<<<cdiv(48L * KP, T), T, 0, stream>>>(Wl3, wb, 47, 48, 128, KP);
  gemm_bf16_wmma<3><<<gemm_blk, T, 0, stream>>>(xb, wb, bl3, lin, N, 47);
  alpha_kernel<<<nh_blk, T, 0, stream>>>(h, att3_src, att3_dst, asrc, adst, N, OUT);
  fill_kernel<<<nh_blk, T, 0, stream>>>(mxb, -1e30f, N * HEADS);
  fill_kernel<<<nh_blk, T, 0, stream>>>(den, 0.f, N * HEADS);
  fill_kernel<<<cdiv((long)N * 188, T), T, 0, stream>>>(agg, 0.f, N * 188);
  edge_max_kernel<<<edg_blk, T, 0, stream>>>(ei, E, ET, asrc, adst, mxb);
  edge_sum_kernel<<<edg_blk, T, 0, stream>>>(ei, E, ET, asrc, adst, mxb, den);
  edge_agg_kernel<<<agg_blk, T, 0, stream>>>(ei, E, ET, asrc, adst, mxb, den, h, agg, OUT);
  final_kernel<<<cdiv(N, T), T, 0, stream>>>(agg, b3, lin, out, N, OUT);
}
